// RotarySelfAttention_5377299054787
// MI455X (gfx1250) — compile-verified, matrix operand order checked by reading, not run on hardware
//
#include <hip/hip_runtime.h>
#include <hip/hip_bf16.h>

#define BB 4
#define LL 2048
#define EE 2048
#define HH 16
#define HD 128
#define MM (BB * LL) // 8192 rows

typedef __attribute__((ext_vector_type(16))) __bf16      v16bf;
typedef __attribute__((ext_vector_type(8)))  float       v8f;
typedef __attribute__((ext_vector_type(4)))  unsigned int v4u;
typedef __attribute__((ext_vector_type(4)))  float       v4f;

struct Frag2 { v4u lo, hi; };

#define WMMA_BF16(a, b, c) \
  __builtin_amdgcn_wmma_f32_16x16x32_bf16(false, (a), false, (b), (short)0, (c), false, false)

__device__ inline unsigned short f2bf(float f) {
  unsigned int u = __builtin_bit_cast(unsigned int, f);
  u += 0x7FFFu + ((u >> 16) & 1u);
  return (unsigned short)(u >> 16);
}
__device__ inline float bf2f(unsigned short s) {
  unsigned int u = ((unsigned int)s) << 16;
  return __builtin_bit_cast(float, u);
}

// 16x32 bf16 fragment (A layout; also B when the source is N-major):
// lane r=lane&15 -> row r; lanes 0-15 hold K 0..7 & 16..23, lanes 16-31 hold K 8..15 & 24..31.
__device__ __attribute__((always_inline)) inline v16bf load_frag(const unsigned short* base, int ld) {
  int lane = threadIdx.x & 31;
  int r = lane & 15, hi = lane >> 4;
  const unsigned short* p = base + (size_t)r * ld + hi * 8;
  Frag2 f;
  f.lo = *(const v4u*)(p);
  f.hi = *(const v4u*)(p + 16);
  return __builtin_bit_cast(v16bf, f);
}

// Transpose + convert one EExEE fp32 weight (K x N) -> bf16 Wt (N x K).
__global__ __launch_bounds__(256) void wprep_kernel(const float* __restrict__ W,
                                                    unsigned short* __restrict__ Wt) {
  __shared__ float tile[32][33];
  const int t = threadIdx.x;
  const int n0 = blockIdx.x * 32, k0 = blockIdx.y * 32;
  const int a = t >> 5, b = t & 31; // a:0..7, b:0..31
  for (int j = 0; j < 4; ++j)
    tile[a + j * 8][b] = W[(size_t)(k0 + a + j * 8) * EE + n0 + b]; // coalesced read
  __syncthreads();
  for (int j = 0; j < 4; ++j)
    Wt[(size_t)(n0 + a + j * 8) * EE + k0 + b] = f2bf(tile[b][a + j * 8]); // coalesced write
}

// C = A(MxK) @ B(KxN). A: fp32 (ABF=0) or bf16 (ABF=1) row-major; Bt: bf16 N x K (pre-transposed).
// Block tile 64x128, 8 waves as 2(m) x 4(n), each wave 32x32 = 4 WMMAs per K-step.
// MODE 0: fp32 row-major out; 1: bf16 row-major out; 2: bf16 out transposed to [b][h][d][l].
template <int ABF, int MODE>
__global__ __launch_bounds__(256) void wmma_gemm_kernel(
    const void* __restrict__ Asrc, const unsigned short* __restrict__ Bt,
    void* __restrict__ Cout) {
  __shared__ unsigned short sA[64 * 32];  // 4 KB
  __shared__ unsigned short sB[128 * 32]; // 8 KB, sB[n][k]

  const int t = threadIdx.x;
  const int wave = t >> 5, lane = t & 31;
  const int wm = wave >> 2, wn = wave & 3;
  const int r = lane & 15, hi = lane >> 4;
  const int n0 = blockIdx.x * 128;
  const int m0 = blockIdx.y * 64;

  const float*          Af = (const float*)Asrc;
  const unsigned short* Ab = (const unsigned short*)Asrc;

  v8f acc00 = {}, acc01 = {}, acc10 = {}, acc11 = {};

  const int arow = t >> 2, ac8 = (t & 3) * 8;   // A: 8 bf16 per thread
  const int brow0 = t >> 2, bc8 = (t & 3) * 8;  // B: 2 chunks per thread (rows +0, +64)

  for (int kt = 0; kt < EE / 32; ++kt) {
    // ---- A tile 64x32 -> LDS bf16 ----
    {
      size_t off = (size_t)(m0 + arow) * EE + (size_t)kt * 32 + ac8;
      v4u d;
      if (ABF) {
        d = *(const v4u*)(Ab + off);
        if (kt + 1 < EE / 32) __builtin_prefetch(Ab + off + 32, 0, 0);
      } else {
        v4f f0 = *(const v4f*)(Af + off);
        v4f f1 = *(const v4f*)(Af + off + 4);
        if (kt + 1 < EE / 32) __builtin_prefetch(Af + off + 32, 0, 0);
        d.x = (unsigned)f2bf(f0.x) | ((unsigned)f2bf(f0.y) << 16);
        d.y = (unsigned)f2bf(f0.z) | ((unsigned)f2bf(f0.w) << 16);
        d.z = (unsigned)f2bf(f1.x) | ((unsigned)f2bf(f1.y) << 16);
        d.w = (unsigned)f2bf(f1.z) | ((unsigned)f2bf(f1.w) << 16);
      }
      *(v4u*)(&sA[arow * 32 + ac8]) = d;
    }
    // ---- Bt tile 128(n) x 32(k) bf16 -> LDS (vectorized) ----
    {
      size_t off0 = (size_t)(n0 + brow0) * EE + (size_t)kt * 32 + bc8;
      size_t off1 = (size_t)(n0 + brow0 + 64) * EE + (size_t)kt * 32 + bc8;
      v4u d0 = *(const v4u*)(Bt + off0);
      v4u d1 = *(const v4u*)(Bt + off1);
      if (kt + 1 < EE / 32) {
        __builtin_prefetch(Bt + off0 + 32, 0, 0);
        __builtin_prefetch(Bt + off1 + 32, 0, 0);
      }
      *(v4u*)(&sB[brow0 * 32 + bc8]) = d0;
      *(v4u*)(&sB[(brow0 + 64) * 32 + bc8]) = d1;
    }
    __syncthreads();
    v16bf a0 = load_frag(&sA[(wm * 32) * 32], 32);
    v16bf a1 = load_frag(&sA[(wm * 32 + 16) * 32], 32);
    v16bf b0 = load_frag(&sB[(wn * 32) * 32], 32);
    v16bf b1 = load_frag(&sB[(wn * 32 + 16) * 32], 32);
    acc00 = WMMA_BF16(a0, b0, acc00);
    acc01 = WMMA_BF16(a0, b1, acc01);
    acc10 = WMMA_BF16(a1, b0, acc10);
    acc11 = WMMA_BF16(a1, b1, acc11);
    __syncthreads();
  }

  v8f* accs[4] = {&acc00, &acc01, &acc10, &acc11};
  for (int mi = 0; mi < 2; ++mi)
    for (int ni = 0; ni < 2; ++ni) {
      v8f acc = *accs[mi * 2 + ni];
      for (int i = 0; i < 8; ++i) {
        int row = m0 + wm * 32 + mi * 16 + i + hi * 8;
        int col = n0 + wn * 32 + ni * 16 + r;
        float v = acc[i];
        if (MODE == 0) {
          ((float*)Cout)[(size_t)row * EE + col] = v;
        } else if (MODE == 1) {
          ((unsigned short*)Cout)[(size_t)row * EE + col] = f2bf(v);
        } else {
          int bb = row >> 11, l = row & (LL - 1);
          int h = col >> 7, d = col & (HD - 1);
          size_t idx = (((size_t)(bb * HH + h) * HD + d) << 11) + l; // <<11 == *LL
          ((unsigned short*)Cout)[idx] = f2bf(v);
        }
      }
    }
}

// In-place rotary on bf16 Q and K (pair of bf16 = one u32).
__global__ __launch_bounds__(256) void rotary_kernel(
    unsigned short* __restrict__ q, unsigned short* __restrict__ k,
    const float* __restrict__ fc, const float* __restrict__ fs) {
  int p = blockIdx.x * 256 + threadIdx.x; // over MM * HH * 64
  int row = p / (HH * 64);
  int rem = p - row * (HH * 64);
  int h = rem >> 6;
  int j = rem & 63;
  int l = row & (LL - 1);
  size_t base = (size_t)row * EE + h * HD + 2 * j;
  float c = fc[l * 64 + j], s = fs[l * 64 + j];

  unsigned int uq = *(unsigned int*)(q + base);
  float qr = bf2f((unsigned short)(uq & 0xFFFF));
  float qi = bf2f((unsigned short)(uq >> 16));
  *(unsigned int*)(q + base) =
      (unsigned)f2bf(qr * c - qi * s) | ((unsigned)f2bf(qr * s + qi * c) << 16);

  unsigned int uk = *(unsigned int*)(k + base);
  float kr = bf2f((unsigned short)(uk & 0xFFFF));
  float ki = bf2f((unsigned short)(uk >> 16));
  *(unsigned int*)(k + base) =
      (unsigned)f2bf(kr * c - ki * s) | ((unsigned)f2bf(kr * s + ki * c) << 16);
}

// Flash attention: 1 wave per 16-query tile, 32-key steps, online softmax.
__global__ __launch_bounds__(256) void attn_kernel(
    const unsigned short* __restrict__ q, const unsigned short* __restrict__ k,
    const unsigned short* __restrict__ vt, const int* __restrict__ mask,
    unsigned short* __restrict__ aout) {
  __shared__ unsigned short sP[8][16 * 32];
  const int t = threadIdx.x, wave = t >> 5, lane = t & 31;
  const int r = lane & 15, hi = lane >> 4;
  const int q0 = blockIdx.x * 128 + wave * 16;
  const int bh = blockIdx.y, b = bh >> 4, h = bh & 15;
  const float scale = 0.08838834764831845f; // 1/sqrt(128)

  v16bf aq[4];
  for (int kk = 0; kk < 4; ++kk)
    aq[kk] = load_frag(q + (size_t)(b * LL + q0) * EE + h * HD + kk * 32, EE);

  float m_i[8], l_i[8];
  v8f accO[8];
  for (int i = 0; i < 8; ++i) { m_i[i] = -1e30f; l_i[i] = 0.f; v8f z = {}; accO[i] = z; }

  for (int kb = 0; kb < LL / 32; ++kb) {
    int k0 = kb * 32;
    v8f s0 = {}, s1 = {};
    for (int kk = 0; kk < 4; ++kk) {
      v16bf b0 = load_frag(k + (size_t)(b * LL + k0) * EE + h * HD + kk * 32, EE);
      s0 = WMMA_BF16(aq[kk], b0, s0);
      v16bf b1 = load_frag(k + (size_t)(b * LL + k0 + 16) * EE + h * HD + kk * 32, EE);
      s1 = WMMA_BF16(aq[kk], b1, s1);
    }
    if (kb + 1 < LL / 32) { // prefetch next K/V tiles
      __builtin_prefetch(k + (size_t)(b * LL + k0 + 32 + r) * EE + h * HD, 0, 0);
      __builtin_prefetch(vt + ((size_t)(b * HH + h) * HD + r * 8) * LL + k0 + 32, 0, 0);
    }
    float mb0 = (mask[b * LL + k0 + r] == 1) ? 0.f : -1e30f;
    float mb1 = (mask[b * LL + k0 + 16 + r] == 1) ? 0.f : -1e30f;

    float p0[8], p1[8], alpha[8];
    for (int i = 0; i < 8; ++i) {
      float a0 = s0[i] * scale + mb0;
      float a1 = s1[i] * scale + mb1;
      float rm = fmaxf(a0, a1);
      rm = fmaxf(rm, __shfl_xor(rm, 1));
      rm = fmaxf(rm, __shfl_xor(rm, 2));
      rm = fmaxf(rm, __shfl_xor(rm, 4));
      rm = fmaxf(rm, __shfl_xor(rm, 8));
      float mn = fmaxf(m_i[i], rm);
      alpha[i] = __expf(m_i[i] - mn);
      p0[i] = __expf(a0 - mn);
      p1[i] = __expf(a1 - mn);
      float rs = p0[i] + p1[i];
      rs += __shfl_xor(rs, 1);
      rs += __shfl_xor(rs, 2);
      rs += __shfl_xor(rs, 4);
      rs += __shfl_xor(rs, 8);
      l_i[i] = l_i[i] * alpha[i] + rs;
      m_i[i] = mn;
    }
    for (int dt = 0; dt < 8; ++dt)
      for (int i = 0; i < 8; ++i) accO[dt][i] *= alpha[i];

    // Re-stripe P (C-layout) through per-wave LDS tile into A-fragment layout.
    unsigned short* pb = sP[wave];
    for (int i = 0; i < 8; ++i) {
      int rr = i + hi * 8;
      pb[rr * 32 + r]      = f2bf(p0[i]);
      pb[rr * 32 + 16 + r] = f2bf(p1[i]);
    }
    asm volatile("s_wait_dscnt 0" ::: "memory");
    v16bf ap = load_frag(pb, 32);

    for (int dt = 0; dt < 8; ++dt) {
      v16bf bv = load_frag(vt + ((size_t)(b * HH + h) * HD + dt * 16) * LL + k0, LL);
      accO[dt] = WMMA_BF16(ap, bv, accO[dt]);
    }
  }

  for (int i = 0; i < 8; ++i) l_i[i] = 1.f / l_i[i];
  for (int dt = 0; dt < 8; ++dt)
    for (int i = 0; i < 8; ++i) {
      int row = q0 + i + hi * 8;
      int col = h * HD + dt * 16 + r;
      aout[(size_t)(b * LL + row) * EE + col] = f2bf(accO[dt][i] * l_i[i]);
    }
}

extern "C" void kernel_launch(void* const* d_in, const int* in_sizes, int n_in,
                              void* d_out, int out_size, void* d_ws, size_t ws_size,
                              hipStream_t stream) {
  const float* x   = (const float*)d_in[0];
  const float* fc  = (const float*)d_in[1];
  const float* fs  = (const float*)d_in[2];
  const int*   msk = (const int*)d_in[3];
  const float* Wq  = (const float*)d_in[4];
  const float* Wk  = (const float*)d_in[5];
  const float* Wv  = (const float*)d_in[6];
  const float* Wo  = (const float*)d_in[7];

  size_t SZ  = (size_t)MM * EE; // activation-sized bf16 buffers
  size_t SZW = (size_t)EE * EE; // weight-sized bf16 buffers
  unsigned short* qb  = (unsigned short*)d_ws;
  unsigned short* kb  = qb + SZ;
  unsigned short* vt  = kb + SZ;
  unsigned short* ao  = vt + SZ;
  unsigned short* wqt = ao + SZ;
  unsigned short* wkt = wqt + SZW;
  unsigned short* wvt = wkt + SZW;
  unsigned short* wot = wvt + SZW;

  // Pre-transpose + convert weights (fp32 KxN -> bf16 NxK) once.
  dim3 gw(EE / 32, EE / 32);
  wprep_kernel<<<gw, 256, 0, stream>>>(Wq, wqt);
  wprep_kernel<<<gw, 256, 0, stream>>>(Wk, wkt);
  wprep_kernel<<<gw, 256, 0, stream>>>(Wv, wvt);
  wprep_kernel<<<gw, 256, 0, stream>>>(Wo, wot);

  dim3 gg(EE / 128, MM / 64);
  wmma_gemm_kernel<0, 1><<<gg, 256, 0, stream>>>(x, wqt, qb);
  wmma_gemm_kernel<0, 1><<<gg, 256, 0, stream>>>(x, wkt, kb);
  wmma_gemm_kernel<0, 2><<<gg, 256, 0, stream>>>(x, wvt, vt); // V stored [b][h][d][l]

  int pairs = MM * HH * (HD / 2);
  rotary_kernel<<<pairs / 256, 256, 0, stream>>>(qb, kb, fc, fs);

  attn_kernel<<<dim3(LL / 128, BB * HH), 256, 0, stream>>>(qb, kb, vt, msk, ao);

  wmma_gemm_kernel<1, 0><<<gg, 256, 0, stream>>>(ao, wot, d_out);
}
